// SimpleHandwritingPredictionModel_16346645529346
// MI455X (gfx1250) — matrix operand order repaired and checked by reading, not compiled, not run on hardware
//
#include <hip/hip_runtime.h>
#include <stdint.h>

// ---------------------------------------------------------------------------
// Peephole LSTM (B=64, T=700, F=3, U=400, L=3) for MI455X / gfx1250.
//   - recurrent GEMM: 50 WGs x 8 units, Wh slice LDS-resident (26.6KB bf16)
//   - v_wmma_f32_16x16x32_bf16 for all GEMMs, f32 accumulate
//   - h broadcast streamed into LDS via global_load_async_to_lds_b128
//   - per-step grid sync via agent-scope atomics + s_sleep
// ---------------------------------------------------------------------------

using bf16 = __bf16;
typedef bf16  v8bf  __attribute__((ext_vector_type(8)));
typedef bf16  v16bf __attribute__((ext_vector_type(16)));
typedef float v8f   __attribute__((ext_vector_type(8)));

#define BATCH 64
#define TLEN  700
#define NROW  (BATCH * TLEN)   // 44800
#define UDIM  400
#define GDIM  1600             // 4*U
#define KPAD  416              // 13 chunks of 32
#define NWG   50               // workgroups in recurrent kernel
#define UPW   8                // units per WG  (32 gate columns)

__device__ inline v16bf cat16(v8bf lo, v8bf hi) {
  return __builtin_shufflevector(lo, hi, 0,1,2,3,4,5,6,7,8,9,10,11,12,13,14,15);
}
__device__ inline v8f wmma_bf16(v16bf a, v16bf b, v8f c) {
  return __builtin_amdgcn_wmma_f32_16x16x32_bf16(false, a, false, b, (short)0, c,
                                                 false, false);
}
__device__ inline float sigm(float x) { return 1.f / (1.f + __expf(-x)); }
__device__ inline float tanh_f(float x) { return 1.f - 2.f / (__expf(2.f * x) + 1.f); }

// ---------------------------------------------------------------------------
// Weight prep: WxT[n][k] = Wx[k][n] (bf16, k padded to 416 with zeros)
// ---------------------------------------------------------------------------
__global__ __launch_bounds__(256) void prep_wxt_kernel(const float* __restrict__ Wx,
                                                       bf16* __restrict__ WxT) {
  const int n = blockIdx.x;                     // 0..1599
  for (int k = threadIdx.x; k < KPAD; k += 256)
    WxT[(size_t)n * KPAD + k] = (k < UDIM) ? (bf16)Wx[(size_t)k * GDIM + n] : (bf16)0.f;
}

// WhP[l][wg][c][k], c = gate*8 + lu, value = Wh[l][k][gate*400 + wg*8 + lu]
__global__ __launch_bounds__(256) void prep_whp_kernel(const float* __restrict__ Wh,
                                                       bf16* __restrict__ WhP) {
  const int wg = blockIdx.x;                    // 0..49
  const int l  = blockIdx.y;                    // 0..2
  bf16* dst = WhP + (((size_t)l * NWG + wg) * 32) * KPAD;
  for (int idx = threadIdx.x; idx < 32 * KPAD; idx += 256) {
    const int c = idx / KPAD, k = idx % KPAD;
    const int gate = c >> 3, lu = c & 7;
    float v = 0.f;
    if (k < UDIM) v = Wh[((size_t)l * UDIM + k) * GDIM + gate * UDIM + wg * UPW + lu];
    dst[idx] = (bf16)v;
  }
}

// ---------------------------------------------------------------------------
// Layer-0 input projection (K=3): xp[row][c] = b0[c] + sum_d x[b][t][d]*Wx0[d][c]
// row = t*64 + b
// ---------------------------------------------------------------------------
__global__ __launch_bounds__(256) void xp0_kernel(const float* __restrict__ x,
                                                  const float* __restrict__ Wx0,
                                                  const float* __restrict__ b0,
                                                  float* __restrict__ xp) {
  const int row = blockIdx.x;
  const int t = row >> 6, b = row & 63;
  const float* xr = x + ((size_t)b * TLEN + t) * 3;
  const float x0 = xr[0], x1 = xr[1], x2 = xr[2];
  float* dst = xp + (size_t)row * GDIM;
  for (int c = threadIdx.x; c < GDIM; c += 256)
    dst[c] = b0[c] + x0 * Wx0[c] + x1 * Wx0[GDIM + c] + x2 * Wx0[2 * GDIM + c];
}

// ---------------------------------------------------------------------------
// Batch input GEMM for layers 1,2:  xp = hseq(bf16,[44800][400]) @ Wx + bias
// grid (700, 25), block 128 (4 waves). Wave w -> M-tile w, 4 N-tiles. WMMA.
// ---------------------------------------------------------------------------
__global__ __launch_bounds__(128) void xp_gemm_kernel(const bf16* __restrict__ Aseq,
                                                      const bf16* __restrict__ WxT,
                                                      const float* __restrict__ bias,
                                                      float* __restrict__ xp) {
  const int lane = threadIdx.x & 31;
  const int mt = threadIdx.x >> 5;
  const int rbase = blockIdx.x * 64 + mt * 16;
  const int cbase = blockIdx.y * 64;
  const int m = rbase + (lane & 15);
  const int aoff = (lane >> 4) << 3;     // 0 / 8
  const int boff = (lane >> 4) << 3;     // 0 / 8
  v8f acc[4] = {{}, {}, {}, {}};
  for (int kc = 0; kc < 13; ++kc) {
    const int o0 = kc * 32 + aoff;
    const int o1 = o0 + 16;
    v8bf alo = {}, ahi = {};
    if (o0 < UDIM) alo = *(const v8bf*)(Aseq + (size_t)m * UDIM + o0);
    if (o1 < UDIM) ahi = *(const v8bf*)(Aseq + (size_t)m * UDIM + o1);
    const v16bf a = cat16(alo, ahi);
    const int kb = kc * 32 + boff;
#pragma unroll
    for (int nt = 0; nt < 4; ++nt) {
      const int cc = cbase + nt * 16 + (lane & 15);
      const bf16* bp = WxT + (size_t)cc * KPAD + kb;
      const v16bf bfrag = cat16(*(const v8bf*)bp, *(const v8bf*)(bp + 16));
      acc[nt] = wmma_bf16(a, bfrag, acc[nt]);
    }
  }
  const int mofs = (lane >> 4) << 3;
#pragma unroll
  for (int nt = 0; nt < 4; ++nt) {
    const int cc = cbase + nt * 16 + (lane & 15);
    const float bb = bias[cc];
#pragma unroll
    for (int v = 0; v < 8; ++v)
      xp[(size_t)(rbase + mofs + v) * GDIM + cc] = acc[nt][v] + bb;
  }
}

// ---------------------------------------------------------------------------
// Persistent recurrent kernel: one layer, 50 WGs x 256 threads (8 waves).
// WG j owns units [8j, 8j+8) -> 32 gate columns. Wh slice in LDS.
// Per step: z = h @ WhSlice (WMMA, h async-streamed to LDS), gates, grid sync.
// ---------------------------------------------------------------------------
__global__ __launch_bounds__(256) void lstm_layer_kernel(
    const float* __restrict__ xp,      // [T*64][1600]
    const bf16* __restrict__ whp,      // [NWG][32][416] for this layer
    const float* __restrict__ pi, const float* __restrict__ pf,
    const float* __restrict__ po,
    bf16* __restrict__ hbuf,           // [64][416], zeroed before launch
    bf16* __restrict__ hseq,           // [T*64][400]
    unsigned* __restrict__ syncc,      // zeroed before launch
    int T) {
  __shared__ alignas(16) bf16 sWh[32][KPAD];     // 26624 B
  __shared__ alignas(16) bf16 sH[2][64][32];     // 8192 B
  __shared__ alignas(16) float sZ[64][32];       // 8192 B

  const int tid = threadIdx.x;
  const int wg = blockIdx.x;
  const int lane = tid & 31;
  const int wv = tid >> 5;      // 0..7
  const int mt = wv >> 1;       // M-tile 0..3
  const int nt = wv & 1;        // N-tile 0..1

  // Load this WG's Wh slice into LDS (once).
  const bf16* wsrc = whp + (size_t)wg * 32 * KPAD;
  for (int i = tid; i < 32 * KPAD; i += 256) (&sWh[0][0])[i] = wsrc[i];

  // Elementwise mapping: thread -> (b, u0..u0+1); cell state in registers.
  const int b = tid >> 2;
  const int u0 = (tid & 3) * 2, u1 = u0 + 1;
  const int gu0 = wg * UPW + u0, gu1 = wg * UPW + u1;
  float c0 = 0.f, c1 = 0.f;
  const float pi0 = pi[gu0], pf0 = pf[gu0], po0 = po[gu0];
  const float pi1 = pi[gu1], pf1 = pf[gu1], po1 = po[gu1];

  // Async h streaming: each thread copies one 16B segment per 32-col chunk.
  const int hrow = tid >> 2, hseg = tid & 3;
  const unsigned lds0 = (unsigned)(uintptr_t)(&sH[0][0][0]) + (unsigned)tid * 16u;
  const unsigned lds1 = (unsigned)(uintptr_t)(&sH[1][0][0]) + (unsigned)tid * 16u;
  const unsigned gofs = (unsigned)(hrow * (KPAD * 2) + hseg * 16);
  const uint64_t hb64 = (uint64_t)(uintptr_t)hbuf;

  const int amo = (lane >> 4) << 3;   // A K-offset 0/8
  const int bmo = (lane >> 4) << 3;   // B K-offset 0/8
  const int am = mt * 16 + (lane & 15);
  const int bc = nt * 16 + (lane & 15);

  __syncthreads();

  for (int t = 0; t < T; ++t) {
    v8f acc = {};
    // pre-issue chunk 0 -> buffer 0
    {
      unsigned voff = gofs;
      asm volatile("global_load_async_to_lds_b128 %0, %1, %2 offset:0"
                   :: "v"(lds0), "v"(voff), "s"(hb64) : "memory");
    }
    for (int kc = 0; kc < 13; ++kc) {
      const int cur = kc & 1;
      if (kc < 12) {
        __syncthreads();  // prior reads of the other buffer are done
        unsigned voff = gofs + (unsigned)(kc + 1) * 64u;
        unsigned dst = ((kc + 1) & 1) ? lds1 : lds0;
        asm volatile("global_load_async_to_lds_b128 %0, %1, %2 offset:0"
                     :: "v"(dst), "v"(voff), "s"(hb64) : "memory");
        asm volatile("s_wait_asynccnt 1" ::: "memory");
      } else {
        asm volatile("s_wait_asynccnt 0" ::: "memory");
      }
      __syncthreads();  // chunk kc resident in LDS for all waves

      const v16bf a = cat16(*(const v8bf*)(&sH[cur][am][amo]),
                            *(const v8bf*)(&sH[cur][am][amo + 16]));
      const int kb = kc * 32 + bmo;
      const v16bf bm = cat16(*(const v8bf*)(&sWh[bc][kb]),
                             *(const v8bf*)(&sWh[bc][kb + 16]));
      acc = wmma_bf16(a, bm, acc);
    }

    // z tile -> LDS
    {
      const int n = nt * 16 + (lane & 15);
      const int mb = mt * 16 + ((lane >> 4) << 3);
#pragma unroll
      for (int v = 0; v < 8; ++v) sZ[mb + v][n] = acc[v];
    }
    __syncthreads();

    // gates + state update (2 cells per thread)
    {
      const float* xr = xp + (size_t)(t * BATCH + b) * GDIM;
      float zi = sZ[b][u0]      + xr[gu0];
      float zf = sZ[b][8 + u0]  + xr[UDIM + gu0];
      float zg = sZ[b][16 + u0] + xr[2 * UDIM + gu0];
      float zo = sZ[b][24 + u0] + xr[3 * UDIM + gu0];
      float ig = sigm(zi + pi0 * c0);
      float fg = sigm(zf + pf0 * c0);
      float cn = fg * c0 + ig * tanh_f(zg);
      float og = sigm(zo + po0 * cn);
      float hn = og * tanh_f(cn);
      c0 = cn;
      bf16 hv = (bf16)hn;
      hbuf[b * KPAD + gu0] = hv;
      hseq[(size_t)(t * BATCH + b) * UDIM + gu0] = hv;

      zi = sZ[b][u1]      + xr[gu1];
      zf = sZ[b][8 + u1]  + xr[UDIM + gu1];
      zg = sZ[b][16 + u1] + xr[2 * UDIM + gu1];
      zo = sZ[b][24 + u1] + xr[3 * UDIM + gu1];
      ig = sigm(zi + pi1 * c1);
      fg = sigm(zf + pf1 * c1);
      cn = fg * c1 + ig * tanh_f(zg);
      og = sigm(zo + po1 * cn);
      hn = og * tanh_f(cn);
      c1 = cn;
      hv = (bf16)hn;
      hbuf[b * KPAD + gu1] = hv;
      hseq[(size_t)(t * BATCH + b) * UDIM + gu1] = hv;
    }

    // grid-wide step barrier (monotonic counter, reset by memset each launch)
    __threadfence();           // release my hbuf writes to agent scope
    __syncthreads();
    if (tid == 0) {
      __hip_atomic_fetch_add(syncc, 1u, __ATOMIC_RELAXED, __HIP_MEMORY_SCOPE_AGENT);
      const unsigned target = (unsigned)NWG * (unsigned)(t + 1);
      while (__hip_atomic_load(syncc, __ATOMIC_RELAXED, __HIP_MEMORY_SCOPE_AGENT) < target)
        __builtin_amdgcn_s_sleep(1);
    }
    __syncthreads();
    __threadfence();           // acquire: invalidate before reading new h
  }
}

// ---------------------------------------------------------------------------
// Output projection: out[b][t][f] = bd[f] + sum_u hseq2[t*64+b][u] * Wd[u][f]
// ---------------------------------------------------------------------------
__global__ __launch_bounds__(256) void out_proj_kernel(const bf16* __restrict__ hseq,
                                                       const float* __restrict__ Wd,
                                                       const float* __restrict__ bd,
                                                       float* __restrict__ out) {
  const int row = blockIdx.x * 256 + threadIdx.x;
  if (row >= NROW) return;
  const int t = row >> 6, b = row & 63;
  const bf16* h = hseq + (size_t)row * UDIM;
  float a0 = 0.f, a1 = 0.f, a2 = 0.f;
  for (int u = 0; u < UDIM; ++u) {
    const float hv = (float)h[u];
    a0 += hv * Wd[u * 3 + 0];
    a1 += hv * Wd[u * 3 + 1];
    a2 += hv * Wd[u * 3 + 2];
  }
  float* o = out + ((size_t)b * TLEN + t) * 3;
  o[0] = a0 + bd[0]; o[1] = a1 + bd[1]; o[2] = a2 + bd[2];
}

// ---------------------------------------------------------------------------
extern "C" void kernel_launch(void* const* d_in, const int* in_sizes, int n_in,
                              void* d_out, int out_size, void* d_ws, size_t ws_size,
                              hipStream_t stream) {
  const float* x   = (const float*)d_in[0];
  const float* Wx0 = (const float*)d_in[1];
  const float* Wx1 = (const float*)d_in[2];
  const float* Wx2 = (const float*)d_in[3];
  const float* Wh  = (const float*)d_in[4];
  const float* bv  = (const float*)d_in[5];
  const float* pi  = (const float*)d_in[6];
  const float* pf  = (const float*)d_in[7];
  const float* po  = (const float*)d_in[8];
  const float* Wd  = (const float*)d_in[9];
  const float* bd  = (const float*)d_in[10];
  float* out = (float*)d_out;

  char* ws = (char*)d_ws;
  size_t off = 0;
  auto carve = [&](size_t bytes) -> char* {
    char* p = ws + off;
    off = (off + bytes + 255) & ~(size_t)255;
    return p;
  };
  float* xp       = (float*)carve((size_t)NROW * GDIM * 4);       // 286.7 MB
  bf16*  hseqA    = (bf16*)carve((size_t)NROW * UDIM * 2);        // 35.8 MB
  bf16*  hseqB    = (bf16*)carve((size_t)NROW * UDIM * 2);        // 35.8 MB
  bf16*  wxt1     = (bf16*)carve((size_t)GDIM * KPAD * 2);
  bf16*  wxt2     = (bf16*)carve((size_t)GDIM * KPAD * 2);
  bf16*  whp      = (bf16*)carve((size_t)3 * NWG * 32 * KPAD * 2);
  bf16*  hbuf     = (bf16*)carve((size_t)BATCH * KPAD * 2);
  unsigned* syncc = (unsigned*)carve(256);

  prep_wxt_kernel<<<GDIM, 256, 0, stream>>>(Wx1, wxt1);
  prep_wxt_kernel<<<GDIM, 256, 0, stream>>>(Wx2, wxt2);
  prep_whp_kernel<<<dim3(NWG, 3), 256, 0, stream>>>(Wh, whp);
  xp0_kernel<<<NROW, 256, 0, stream>>>(x, Wx0, bv, xp);

  bf16* hin = hseqB;
  bf16* hout = hseqA;
  for (int l = 0; l < 3; ++l) {
    if (l > 0) {
      const bf16* wxt = (l == 1) ? wxt1 : wxt2;
      xp_gemm_kernel<<<dim3(TLEN, GDIM / 64), 128, 0, stream>>>(hin, wxt,
                                                                bv + (size_t)l * GDIM, xp);
    }
    hipMemsetAsync(hbuf, 0, (size_t)BATCH * KPAD * 2, stream);
    hipMemsetAsync(syncc, 0, 256, stream);
    lstm_layer_kernel<<<NWG, 256, 0, stream>>>(
        xp, whp + (size_t)l * NWG * 32 * KPAD, pi + l * UDIM, pf + l * UDIM,
        po + l * UDIM, hbuf, hout, syncc, TLEN);
    bf16* tmp = hin; hin = hout; hout = tmp;
  }
  out_proj_kernel<<<(NROW + 255) / 256, 256, 0, stream>>>(hin, Wd, bd, out);

  (void)in_sizes; (void)n_in; (void)out_size; (void)ws_size;
}
